// FractalSGCNNet_70824010711180
// MI455X (gfx1250) — matrix operand order repaired and checked by reading, not compiled, not run on hardware
//
#include <hip/hip_runtime.h>
#include <hip/hip_bf16.h>

// ---------- types ----------
typedef __bf16 bf16_t;
typedef bf16_t v16bf __attribute__((ext_vector_type(16)));
typedef float  v8f   __attribute__((ext_vector_type(8)));
typedef unsigned int u32x4 __attribute__((ext_vector_type(4)));
typedef float        f32x4 __attribute__((ext_vector_type(4)));

union Frag { u32x4 u[2]; v16bf v; };

// sizes (compile-time constants of the problem)
#define NB 128      // batch
#define NN 64       // nodes
#define NF 2048     // in features
#define NH 512      // hidden
#define NE 512      // edges
#define NPOOL 8
#define NHP (NH / NPOOL)     // 64 pooled channels
#define FCIN (NN * NHP)      // 4096
#define NOUT 4

__device__ __forceinline__ unsigned short f2bf(float f) {
  unsigned int u = __float_as_uint(f);
  unsigned int r = u + 0x7FFFu + ((u >> 16) & 1u);   // round-to-nearest-even
  return (unsigned short)(r >> 16);
}

// ---------------------------------------------------------------------------
// Kernel A: build gcn-normalized dense S [64x64] from edges, compute S^3.
// One block, 256 threads. S3g: [64*64] f32.
// ---------------------------------------------------------------------------
__global__ void prep_kernel(const int* __restrict__ ei, const float* __restrict__ ew,
                            float* __restrict__ S3g) {
  __shared__ float deg[NN];
  __shared__ float dinv[NN];
  __shared__ float S[NN * NN];
  __shared__ float T[NN * NN];
  const int tid = threadIdx.x;

  for (int i = tid; i < NN; i += 256) deg[i] = 0.f;
  for (int i = tid; i < NN * NN; i += 256) S[i] = 0.f;
  __syncthreads();

  // degree over target (col) incl. self loops (weight 1)
  for (int e = tid; e < NE; e += 256) atomicAdd(&deg[ei[NE + e]], ew[e]);
  if (tid < NN) atomicAdd(&deg[tid], 1.0f);
  __syncthreads();
  if (tid < NN) {
    float d = deg[tid];
    dinv[tid] = (d > 0.f) ? rsqrtf(fmaxf(d, 1e-30f)) : 0.f;
  }
  __syncthreads();

  // S[c][r] = dinv[r]*w*dinv[c]; msg flows r -> c (scatter-add on col)
  for (int e = tid; e < NE; e += 256) {
    int r = ei[e], c = ei[NE + e];
    atomicAdd(&S[c * NN + r], dinv[r] * ew[e] * dinv[c]);
  }
  if (tid < NN) atomicAdd(&S[tid * NN + tid], dinv[tid] * dinv[tid]);
  __syncthreads();

  // T = S*S
  for (int i = tid; i < NN * NN; i += 256) {
    int n = i >> 6, m = i & 63;
    float s = 0.f;
    #pragma unroll 8
    for (int p = 0; p < NN; p++) s += S[n * NN + p] * S[p * NN + m];
    T[i] = s;
  }
  __syncthreads();
  // S3 = T*S
  for (int i = tid; i < NN * NN; i += 256) {
    int n = i >> 6, m = i & 63;
    float s = 0.f;
    #pragma unroll 8
    for (int p = 0; p < NN; p++) s += T[n * NN + p] * S[p * NN + m];
    S3g[i] = s;
  }
}

// ---------------------------------------------------------------------------
// Kernel B: per-row inverse L2 norm of x, rows = B*N, each of 2048 floats.
// One wave per row, 8 waves per block. invn: [8192] f32.
// ---------------------------------------------------------------------------
__global__ void rownorm_kernel(const float* __restrict__ x, float* __restrict__ invn) {
  const int row = blockIdx.x * 8 + (threadIdx.x >> 5);
  const int lane = threadIdx.x & 31;
  const f32x4* p = (const f32x4*)(x + (size_t)row * NF);
  float ss = 0.f;
  #pragma unroll 4
  for (int i = lane; i < NF / 4; i += 32) {
    f32x4 v = p[i];
    ss += v.x * v.x + v.y * v.y + v.z * v.z + v.w * v.w;
  }
  #pragma unroll
  for (int m = 16; m; m >>= 1) ss += __shfl_xor(ss, m, 32);
  if (lane == 0) invn[row] = 1.0f / fmaxf(sqrtf(ss), 1e-12f);
}

// ---------------------------------------------------------------------------
// Kernel C: y[b,n,f] = sum_m S3[n][m] * x[b,m,f] * invn[b,m]  -> bf16 A matrix
// grid (B, F/64), 256 threads. A: [8192][2048] bf16 (row-major).
// ---------------------------------------------------------------------------
__global__ void prop_kernel(const float* __restrict__ x, const float* __restrict__ invn,
                            const float* __restrict__ S3, unsigned short* __restrict__ A) {
  __shared__ float xs[NN * 64];
  __shared__ float s3s[NN * NN];
  const int b = blockIdx.x;
  const int f0 = blockIdx.y * 64;
  const int tid = threadIdx.x;

  for (int i = tid; i < NN * NN; i += 256) s3s[i] = S3[i];
  for (int i = tid; i < NN * 64; i += 256) {
    int m = i >> 6, fl = i & 63;
    xs[i] = x[((size_t)b * NN + m) * NF + f0 + fl] * invn[b * NN + m];
  }
  __syncthreads();

  const int f = tid & 63;
  const int ng = tid >> 6;            // 0..3, each handles 16 output nodes
  #pragma unroll 4
  for (int i = 0; i < 16; i++) {
    int n = ng * 16 + i;
    float s = 0.f;
    #pragma unroll 8
    for (int m = 0; m < NN; m++) s += s3s[n * NN + m] * xs[m * 64 + f];
    A[((size_t)b * NN + n) * NF + f0 + f] = f2bf(s);
  }
}

// ---------------------------------------------------------------------------
// Kernel D: pack lin_w [512][2048] f32 -> bf16 (same row-major layout)
// ---------------------------------------------------------------------------
__global__ void packw_kernel(const float* __restrict__ w, unsigned short* __restrict__ wb) {
  int idx = blockIdx.x * blockDim.x + threadIdx.x;      // 262144 threads
  for (int i = idx; i < NH * NF; i += 262144) wb[i] = f2bf(w[i]);
}

// ---------------------------------------------------------------------------
// Kernel E: WMMA GEMM  C[8192,512] = A[8192,2048]bf16 x Wb[512,2048]^T bf16
// fused epilogue: +lin_b -> ELU -> BN(scalar) -> avgpool(8 along h)
// block = 256 thr (8 waves as 4Mx2N); wave tile 32M x 64N (2x4 WMMA tiles);
// block tile 128M x 128N. Single fragment set (no spills), loads clustered
// ahead of the 8 WMMAs; cross-iteration overlap via compiler loop rotation
// and >=2 waves/SIMD (__launch_bounds__(256, 2)).
// grid (8192/128, 512/128) = (64, 4).  P: [8192][64] f32 pooled output.
// ---------------------------------------------------------------------------
__global__ void __launch_bounds__(256, 2)
gemm_kernel(const unsigned short* __restrict__ A,
            const unsigned short* __restrict__ Wb,
            const float* __restrict__ lin_b,
            const float* __restrict__ bn_g, const float* __restrict__ bn_b,
            const float* __restrict__ bn_m, const float* __restrict__ bn_v,
            float* __restrict__ P) {
  const int lane  = threadIdx.x & 31;
  const int wave  = threadIdx.x >> 5;
  const int waveM = wave & 3;          // 4 M-waves
  const int waveN = wave >> 2;         // 2 N-waves
  const int mbase = blockIdx.x * 128 + waveM * 32;
  const int nbase = blockIdx.y * 128 + waveN * 64;
  const int mlane = lane & 15;
  const int hi    = lane >> 4;         // 0/1: K half-select per WMMA layout

  const unsigned short* arow0 = A + (size_t)(mbase + mlane) * NF;        // M-tile 0
  const unsigned short* arow1 = arow0 + (size_t)16 * NF;                 // M-tile 1
  const unsigned short* brow  = Wb + (size_t)(nbase + mlane) * NF + 8 * hi;

  v8f acc[2][4] = {};                  // [M-tile][N-tile]

  for (int k0 = 0; k0 < NF; k0 += 32) {
    // cluster all operand loads for this k-step, then 8 back-to-back WMMAs
    Frag a[2], b[4];
    a[0].u[0] = *(const u32x4*)(arow0 + k0 + 8 * hi);
    a[0].u[1] = *(const u32x4*)(arow0 + k0 + 16 + 8 * hi);
    a[1].u[0] = *(const u32x4*)(arow1 + k0 + 8 * hi);
    a[1].u[1] = *(const u32x4*)(arow1 + k0 + 16 + 8 * hi);
    #pragma unroll
    for (int t = 0; t < 4; t++) {
      const unsigned short* bp = brow + (size_t)t * 16 * NF + k0;
      b[t].u[0] = *(const u32x4*)(bp);
      b[t].u[1] = *(const u32x4*)(bp + 16);
    }
    if (k0 + 128 < NF) {
      __builtin_prefetch(arow0 + k0 + 128, 0, 1);
      __builtin_prefetch(arow1 + k0 + 128, 0, 1);
    }
    #pragma unroll
    for (int mt = 0; mt < 2; mt++)
      #pragma unroll
      for (int t = 0; t < 4; t++)
        acc[mt][t] = __builtin_amdgcn_wmma_f32_16x16x32_bf16(
            false, a[mt].v, false, b[t].v, (short)0, acc[mt][t], false, false);
  }

  // epilogue: bias -> ELU -> BN -> pool(8 along h via lane shuffles)
  const float scale = bn_g[0] * rsqrtf(bn_v[0] + 1e-5f);
  const float shift = bn_b[0] - bn_m[0] * scale;

  #pragma unroll
  for (int mt = 0; mt < 2; mt++) {
    const int mrow0 = mbase + mt * 16 + hi * 8;    // rows mrow0..mrow0+7 in this lane
    #pragma unroll
    for (int t = 0; t < 4; t++) {
      const int hcol = nbase + t * 16 + mlane;     // global h of this lane's column
      const float bias = lin_b[hcol];
      #pragma unroll
      for (int r = 0; r < 8; r++) {
        float v = acc[mt][t][r] + bias;
        v = (v > 0.f) ? v : (expf(v) - 1.0f);      // ELU
        v = v * scale + shift;                     // BN (eval, scalar)
        // sum over 8 consecutive h (lanes sharing bits [4:3])
        v += __shfl_xor(v, 1, 32);
        v += __shfl_xor(v, 2, 32);
        v += __shfl_xor(v, 4, 32);
        if ((lane & 7) == 0) {
          P[(size_t)(mrow0 + r) * NHP + (hcol >> 3)] = v * 0.125f;
        }
      }
    }
  }
}

// ---------------------------------------------------------------------------
// Kernel F: out[b,o] = P_flat[b,:] . fc_w[o,:] + fc_b[o]
// one block per b; 256 threads.
// ---------------------------------------------------------------------------
__global__ void fc_kernel(const float* __restrict__ P, const float* __restrict__ fcw,
                          const float* __restrict__ fcb, float* __restrict__ out) {
  const int b = blockIdx.x;
  const int tid = threadIdx.x;
  const float* pb = P + (size_t)b * FCIN;
  float a0 = 0.f, a1 = 0.f, a2 = 0.f, a3 = 0.f;
  for (int j = tid; j < FCIN; j += 256) {
    float v = pb[j];
    a0 += v * fcw[j];
    a1 += v * fcw[FCIN + j];
    a2 += v * fcw[2 * FCIN + j];
    a3 += v * fcw[3 * FCIN + j];
  }
  #pragma unroll
  for (int m = 16; m; m >>= 1) {
    a0 += __shfl_xor(a0, m, 32);
    a1 += __shfl_xor(a1, m, 32);
    a2 += __shfl_xor(a2, m, 32);
    a3 += __shfl_xor(a3, m, 32);
  }
  __shared__ float sm[8][NOUT];
  if ((tid & 31) == 0) {
    int w = tid >> 5;
    sm[w][0] = a0; sm[w][1] = a1; sm[w][2] = a2; sm[w][3] = a3;
  }
  __syncthreads();
  if (tid < NOUT) {
    float s = 0.f;
    #pragma unroll
    for (int w = 0; w < 8; w++) s += sm[w][tid];
    out[b * NOUT + tid] = s + fcb[tid];
  }
}

// ---------------------------------------------------------------------------
// launch
// ---------------------------------------------------------------------------
extern "C" void kernel_launch(void* const* d_in, const int* in_sizes, int n_in,
                              void* d_out, int out_size, void* d_ws, size_t ws_size,
                              hipStream_t stream) {
  const float* x     = (const float*)d_in[0];
  const int*   ei    = (const int*)d_in[1];
  const float* ew    = (const float*)d_in[2];
  const float* lin_w = (const float*)d_in[3];
  const float* lin_b = (const float*)d_in[4];
  const float* bn_g  = (const float*)d_in[5];
  const float* bn_b  = (const float*)d_in[6];
  const float* bn_m  = (const float*)d_in[7];
  const float* bn_v  = (const float*)d_in[8];
  const float* fc_w  = (const float*)d_in[9];
  const float* fc_b  = (const float*)d_in[10];
  float* out = (float*)d_out;

  char* ws = (char*)d_ws;
  unsigned short* Abf = (unsigned short*)(ws);                       // 32 MB
  unsigned short* Wb  = (unsigned short*)(ws + 33554432);            // 2 MB
  float* P    = (float*)(ws + 35651584);                             // 2 MB
  float* invn = (float*)(ws + 37748736);                             // 32 KB
  float* S3   = (float*)(ws + 37781504);                             // 16 KB

  prep_kernel<<<1, 256, 0, stream>>>(ei, ew, S3);
  rownorm_kernel<<<(NB * NN) / 8, 256, 0, stream>>>(x, invn);
  packw_kernel<<<1024, 256, 0, stream>>>(lin_w, Wb);
  prop_kernel<<<dim3(NB, NF / 64), 256, 0, stream>>>(x, invn, S3, Abf);
  gemm_kernel<<<dim3((NB * NN) / 128, NH / 128), 256, 0, stream>>>(
      Abf, Wb, lin_b, bn_g, bn_b, bn_m, bn_v, P);
  fc_kernel<<<NB, 256, 0, stream>>>(P, fc_w, fc_b, out);
}
